// GaussianRenderer_52321291599930
// MI455X (gfx1250) — matrix operand order repaired and checked by reading, not compiled, not run on hardware
//
#include <hip/hip_runtime.h>
#include <hip/hip_bf16.h>
#include <math.h>

typedef __attribute__((ext_vector_type(16))) _Float16 v16h;
typedef __attribute__((ext_vector_type(8)))  float    v8f;

#define GH 256
#define GW 256
#define NG 1024
#define EPSV 1.0e-4f
#define PSTRIDE 12
#define NBATCH (NG / 16)

// CDNA5 async global->LDS path (ASYNCcnt-tracked, no VGPR staging), guarded so
// toolchains without the builtin fall back to load+ds_store.
#if defined(__has_builtin)
#if __has_builtin(__builtin_amdgcn_global_load_async_to_lds_b128) && \
    __has_builtin(__builtin_amdgcn_s_wait_asynccnt)
#define GS_ASYNC_LDS 1
#endif
#endif
#ifndef GS_ASYNC_LDS
#define GS_ASYNC_LDS 0
#endif

#if GS_ASYNC_LDS
// Builtin prototype (from the compiler diagnostic):
//   void __builtin_amdgcn_global_load_async_to_lds_b128(
//       int __attribute__((vector_size(16))) addrspace(1)* src,
//       int __attribute__((vector_size(16))) addrspace(3)* dst,
//       imm int offset, imm int cpol)
typedef int gs_v4i __attribute__((vector_size(16)));
typedef __attribute__((address_space(1))) gs_v4i* gs_as1_v4i;
typedef __attribute__((address_space(3))) gs_v4i* gs_as3_v4i;
#endif

// ---------------------------------------------------------------------------
// Kernel 1: per-gaussian projection + quadratic-form coefficient precompute.
// params[n] = {m2x, m2y, qxx, qxy, qyy, log(opac*norm*valid), r, g, b, depth, 0, 0}
// P(dx,dy) = qxx*dx^2 + qxy*dx*dy + qyy*dy^2   (already includes the -0.5)
// ---------------------------------------------------------------------------
__global__ void gs_preprocess(const float* __restrict__ means3D,
                              const float* __restrict__ covs3d,
                              const float* __restrict__ colors,
                              const float* __restrict__ opac,
                              const float* __restrict__ Km,
                              const float* __restrict__ Rm,
                              const float* __restrict__ tv,
                              float* __restrict__ depths,
                              float* __restrict__ params) {
  int n = blockIdx.x * blockDim.x + threadIdx.x;
  if (n >= NG) return;

  float m0 = means3D[n * 3 + 0];
  float m1 = means3D[n * 3 + 1];
  float m2 = means3D[n * 3 + 2];

  float cam[3];
#pragma unroll
  for (int i = 0; i < 3; ++i)
    cam[i] = Rm[i * 3 + 0] * m0 + Rm[i * 3 + 1] * m1 + Rm[i * 3 + 2] * m2 + tv[i];

  float depth = fmaxf(cam[2], 1.0f);

  float sp[3];
#pragma unroll
  for (int i = 0; i < 3; ++i)
    sp[i] = Km[i * 3 + 0] * cam[0] + Km[i * 3 + 1] * cam[1] + Km[i * 3 + 2] * cam[2];
  float u = sp[0], v = sp[1], z = sp[2];
  float m2x = u / z;
  float m2y = v / z;
  float iz  = 1.0f / z;
  float iz2 = iz * iz;

  // Jacobian exactly as in the reference (note row0/row1 asymmetry).
  float J[2][3];
  J[0][0] = Km[0] * iz - u * iz2;
  J[0][1] = Km[1] * iz - v * iz2;
  J[0][2] = -u * iz2;
  J[1][0] = Km[3] * iz - u * iz2;
  J[1][1] = Km[4] * iz - v * iz2;
  J[1][2] = -v * iz2;

  // covs_cam = R * C * R^T
  const float* C = covs3d + n * 9;
  float M[3][3], Cc[3][3];
#pragma unroll
  for (int i = 0; i < 3; ++i)
#pragma unroll
    for (int j = 0; j < 3; ++j)
      M[i][j] = Rm[i * 3 + 0] * C[0 * 3 + j] + Rm[i * 3 + 1] * C[1 * 3 + j] +
                Rm[i * 3 + 2] * C[2 * 3 + j];
#pragma unroll
  for (int i = 0; i < 3; ++i)
#pragma unroll
    for (int l = 0; l < 3; ++l)
      Cc[i][l] = M[i][0] * Rm[l * 3 + 0] + M[i][1] * Rm[l * 3 + 1] +
                 M[i][2] * Rm[l * 3 + 2];

  // covs2D = J * Cc * J^T  (+ EPS*I)
  float JC[2][3];
#pragma unroll
  for (int i = 0; i < 2; ++i)
#pragma unroll
    for (int j = 0; j < 3; ++j)
      JC[i][j] = J[i][0] * Cc[0][j] + J[i][1] * Cc[1][j] + J[i][2] * Cc[2][j];

  float a = JC[0][0] * J[0][0] + JC[0][1] * J[0][1] + JC[0][2] * J[0][2] + EPSV;
  float b = JC[0][0] * J[1][0] + JC[0][1] * J[1][1] + JC[0][2] * J[1][2];
  float c = JC[1][0] * J[0][0] + JC[1][1] * J[0][1] + JC[1][2] * J[0][2];
  float d = JC[1][0] * J[1][0] + JC[1][1] * J[1][1] + JC[1][2] * J[1][2] + EPSV;

  float det    = a * d - b * c;
  float invdet = 1.0f / det;
  float qxx = -0.5f * d * invdet;
  float qyy = -0.5f * a * invdet;
  float qxy =  0.5f * (b + c) * invdet;      // -0.5 * (i01 + i10)

  float norm  = 0.15915494309189535f * rsqrtf(det);  // 1/(2*pi*sqrt(det))
  bool  valid = (depth > 1.0f) && (depth < 50.0f);
  float coeff = valid ? (opac[n] * norm) : 0.0f;
  float lc    = __logf(coeff);               // coeff==0 -> -inf -> exp()==0

  depths[n] = depth;
  float* p = params + n * PSTRIDE;
  p[0] = m2x; p[1] = m2y; p[2] = qxx; p[3] = qxy; p[4] = qyy; p[5] = lc;
  p[6] = colors[n * 3 + 0]; p[7] = colors[n * 3 + 1]; p[8] = colors[n * 3 + 2];
  p[9] = depth; p[10] = 0.0f; p[11] = 0.0f;
}

// ---------------------------------------------------------------------------
// Kernel 2: single-workgroup bitonic sort of 1024 depths (LDS), then gather
// packed params into depth-sorted order.
// ---------------------------------------------------------------------------
__global__ void gs_sort(const float* __restrict__ depths,
                        const float* __restrict__ params,
                        float* __restrict__ sorted) {
  __shared__ float key[NG];
  __shared__ int   sidx[NG];
  int t = threadIdx.x;
  key[t]  = depths[t];
  sidx[t] = t;
  __syncthreads();

  for (int k = 2; k <= NG; k <<= 1) {
    for (int j = k >> 1; j > 0; j >>= 1) {
      int ixj = t ^ j;
      if (ixj > t) {
        float ka = key[t], kb = key[ixj];
        bool asc = ((t & k) == 0);
        if (asc ? (ka > kb) : (ka < kb)) {
          key[t] = kb; key[ixj] = ka;
          int tmp = sidx[t]; sidx[t] = sidx[ixj]; sidx[ixj] = tmp;
        }
      }
      __syncthreads();
    }
  }

  int src = sidx[t];
#pragma unroll
  for (int q = 0; q < PSTRIDE; ++q)
    sorted[t * PSTRIDE + q] = params[src * PSTRIDE + q];
}

// ---------------------------------------------------------------------------
// Kernel 3: tiled renderer. One wave32 per 4x4 pixel tile, 8 waves per block.
// Per 16-gaussian batch:
//   stage:   async global->LDS (ASYNCcnt) of the 768B packed batch
//   WMMA #1: P(16px x 16g) = Basis(16px x 8) @ Coef(8 x 16g)   [f16 in, f32 acc]
//   VALU:    alpha = exp(P); cross-lane prefix product (width-16 shfl scan)
//            for transmittance; weights -> LDS (wave-local transpose)
//   WMMA #2: rendered(16px x 16ch) += W(16px x 16g) @ Colors(16g x 16ch)
// ---------------------------------------------------------------------------
__global__ void __launch_bounds__(256) gs_render(const float* __restrict__ sorted,
                                                 float* __restrict__ out) {
  __shared__ __align__(16) float sh_g[16 * PSTRIDE];  // shared gaussian batch
  __shared__ float sh_w[8][16 * 16];                  // per-wave weight transpose

  const int tid  = threadIdx.x;
  const int wave = tid >> 5;
  const int lane = tid & 31;
  const int l16  = lane & 15;
  const bool hi  = lane >= 16;
  const int mofs = hi ? 8 : 0;

  const int tile = blockIdx.x * 8 + wave;
  const int tx = tile & 63;
  const int ty = tile >> 6;
  const float cx = (float)(tx * 4) + 1.5f;   // tile center (keeps f16 basis small)
  const float cy = (float)(ty * 4) + 1.5f;

  // A operand: basis [dx^2, dy^2, dx*dy, dx, dy, 1, 0, 0] in tile coords.
  // 16-bit A 16x32 layout: lanes 0-15 hold K=0..7 in halves 0..7 (K=16..23 in
  // halves 8..15, zero); lanes 16-31 hold K=8..15 / 24..31 (all zero here).
  v16h Ab;
#pragma unroll
  for (int i = 0; i < 16; ++i) Ab[i] = (_Float16)0.0f;
  if (!hi) {
    float xr = (float)(l16 & 3) - 1.5f;
    float yr = (float)(l16 >> 2) - 1.5f;
    Ab[0] = (_Float16)(xr * xr);
    Ab[1] = (_Float16)(yr * yr);
    Ab[2] = (_Float16)(xr * yr);
    Ab[3] = (_Float16)xr;
    Ab[4] = (_Float16)yr;
    Ab[5] = (_Float16)1.0f;
  }

  float Tb[8];                 // per-pixel transmittance (C-layout rows)
#pragma unroll
  for (int j = 0; j < 8; ++j) Tb[j] = 1.0f;
  v8f acc = {};                // rendered RGB accumulator (C-layout)

  for (int b = 0; b < NBATCH; ++b) {
    if (lane == 0 && b + 1 < NBATCH)
      __builtin_prefetch(sorted + (b + 1) * 16 * PSTRIDE, 0, 1);  // global_prefetch_b8

    __syncthreads();   // previous batch fully consumed before restaging

#if GS_ASYNC_LDS
    // 768B batch: 48 lanes x 16B, ASYNCcnt-tracked, no VGPR round trip.
    if (tid < 48)
      __builtin_amdgcn_global_load_async_to_lds_b128(
          (gs_as1_v4i)(sorted + b * 16 * PSTRIDE + tid * 4),
          (gs_as3_v4i)(sh_g + tid * 4),
          0, 0);
    __builtin_amdgcn_s_wait_asynccnt(0);   // wave-local completion before barrier
#else
    if (tid < 16 * PSTRIDE) sh_g[tid] = sorted[b * 16 * PSTRIDE + tid];
#endif
    __syncthreads();

    // B operand: coefficient rows k=0..5 for gaussian n = lane (lanes 0-15).
    // 16-bit B 32x16 layout: lanes 0-15 half h<8 = row K=h; lanes 16-31 = K=8..15/24..31
    // (multiplied by zero A halves, left zero).
    v16h Bc;
#pragma unroll
    for (int i = 0; i < 16; ++i) Bc[i] = (_Float16)0.0f;
    if (!hi) {
      const float* g = sh_g + l16 * PSTRIDE;
      float mx = g[0], my = g[1], qxx = g[2], qxy = g[3], qyy = g[4], lc = g[5];
      float ux = mx - cx, uy = my - cy;
      Bc[0] = (_Float16)qxx;
      Bc[1] = (_Float16)qyy;
      Bc[2] = (_Float16)qxy;
      Bc[3] = (_Float16)(-(2.0f * qxx * ux + qxy * uy));
      Bc[4] = (_Float16)(-(2.0f * qyy * uy + qxy * ux));
      Bc[5] = (_Float16)(qxx * ux * ux + qxy * ux * uy + qyy * uy * uy + lc);
    }

    v8f P = {};
    P = __builtin_amdgcn_wmma_f32_16x16x32_f16(false, Ab, false, Bc,
                                               (short)0, P, false, false);

    // C-layout: lane holds gaussian n=l16, pixels m = j + mofs (j = VGPR row).
    // alpha = exp(P); transmittance = prefix product of (1-alpha) over the 16
    // sorted gaussians, which live across the 16-lane group -> shfl scan.
    float* wsh = sh_w[wave];
#pragma unroll
    for (int j = 0; j < 8; ++j) {
      float al = __expf(P[j]);
      float om = 1.0f - al;
#pragma unroll
      for (int dlt = 1; dlt < 16; dlt <<= 1) {
        float up = __shfl_up(om, (unsigned)dlt, 16);
        if (l16 >= dlt) om *= up;            // inclusive prefix product
      }
      float ex = __shfl_up(om, 1u, 16);      // exclusive prefix
      if (l16 == 0) ex = 1.0f;
      float tot = __shfl(om, 15, 16);        // batch transmittance factor
      wsh[(j + mofs) * 16 + l16] = Tb[j] * ex * al;   // weight W[m][n]
      Tb[j] *= tot;
    }

    // Wave-local LDS transpose (DS ops from one wave are in-order):
    // A operand for color WMMA: lanes 0-15 -> W[m=l16][g=0..7] in halves 0..7,
    // lanes 16-31 -> W[m=l16][g=8..15]; halves 8..15 (K=16..31) zero.
    v16h Aw;
#pragma unroll
    for (int h = 0; h < 8; ++h) Aw[h] = (_Float16)wsh[l16 * 16 + mofs + h];
#pragma unroll
    for (int h = 8; h < 16; ++h) Aw[h] = (_Float16)0.0f;

    // B operand: colors, row k = gaussian, column n = channel (n<3 used).
    v16h Bcol;
#pragma unroll
    for (int i = 0; i < 16; ++i) Bcol[i] = (_Float16)0.0f;
    if (l16 < 3) {
#pragma unroll
      for (int h = 0; h < 16; ++h)
        Bcol[h] = (_Float16)sh_g[((h & 7) + mofs) * PSTRIDE + 6 + l16];
    }

    acc = __builtin_amdgcn_wmma_f32_16x16x32_f16(false, Aw, false, Bcol,
                                                 (short)0, acc, false, false);
  }

  // Write out: lane channel = l16 (<3), pixel rows m = j + mofs.
  if (l16 < 3) {
#pragma unroll
    for (int j = 0; j < 8; ++j) {
      int m  = j + mofs;
      int px = tx * 4 + (m & 3);
      int py = ty * 4 + (m >> 2);
      out[(py * GW + px) * 3 + l16] = acc[j];
    }
  }
}

// ---------------------------------------------------------------------------
extern "C" void kernel_launch(void* const* d_in, const int* in_sizes, int n_in,
                              void* d_out, int out_size, void* d_ws, size_t ws_size,
                              hipStream_t stream) {
  (void)in_sizes; (void)n_in; (void)out_size; (void)ws_size;

  const float* means3D = (const float*)d_in[0];
  const float* covs3d  = (const float*)d_in[1];
  const float* colors  = (const float*)d_in[2];
  const float* opac    = (const float*)d_in[3];
  const float* Km      = (const float*)d_in[4];
  const float* Rm      = (const float*)d_in[5];
  const float* tv      = (const float*)d_in[6];

  float* depths  = (float*)d_ws;              // NG floats
  float* params  = depths + NG;               // NG * PSTRIDE floats
  float* sortedp = params + NG * PSTRIDE;     // NG * PSTRIDE floats (16B aligned)

  gs_preprocess<<<NG / 256, 256, 0, stream>>>(means3D, covs3d, colors, opac,
                                              Km, Rm, tv, depths, params);
  gs_sort<<<1, NG, 0, stream>>>(depths, params, sortedp);
  gs_render<<<(GH / 4) * (GW / 4) / 8, 256, 0, stream>>>(sortedp, (float*)d_out);
}
